// LambdaLayer_77395310674187
// MI455X (gfx1250) — matrix operand order repaired and checked by reading, not compiled, not run on hardware
//
#include <hip/hip_runtime.h>
#include <hip/hip_bf16.h>
#include <stdint.h>

typedef __bf16 v16bf __attribute__((ext_vector_type(16)));
typedef float  v8f   __attribute__((ext_vector_type(8)));

#define BATCH 16
#define CIN   256
#define NPIX  1024
#define EPS   1e-5f

// workspace offsets (in floats)
#define OFF_QRAW 0u          // 16*64*1024
#define OFF_KRAW 1048576u    // 16*16*1024
#define OFF_VRAW 1310720u    // 16*32*1024
#define OFF_STAT 1835008u    // sum[112], sumsq[112]
#define OFF_SCSH 1835232u    // scale_q[64], shift_q[64], scale_v[32], shift_v[32]
#define OFF_LAMC 1835424u    // 16*16*32
#define OFF_VB16 1843616u    // bf16 [b][v][m]  : 524288 u16
#define OFF_PT   2105760u    // bf16 [n][k][m]  : 16777216 u16

__device__ __forceinline__ unsigned short f2bf(float f) {
  unsigned int u = __float_as_uint(f);
  u += 0x7FFFu + ((u >> 16) & 1u);   // round-to-nearest-even
  return (unsigned short)(u >> 16);
}

__global__ void zero_stats_kernel(float* ws) {
  unsigned t = threadIdx.x;
  if (t < 224) ws[OFF_STAT + t] = 0.f;
}

// q/k/v 1x1 convs + BN partial statistics
__global__ __launch_bounds__(256) void proj_stats_kernel(
    const float* __restrict__ x, const float* __restrict__ wq,
    const float* __restrict__ wk, const float* __restrict__ wv,
    float* __restrict__ ws) {
  __shared__ float xs[CIN][32];
  __shared__ float st[224];
  const int b = blockIdx.y;
  const int n0 = blockIdx.x * 32;
  const int tid = threadIdx.x;
  for (int i = 0; i < 32; ++i) {
    int idx = tid + i * 256;
    int c = idx >> 5, j = idx & 31;
    xs[c][j] = x[((b * CIN + c) << 10) + n0 + j];
  }
  if (tid < 224) st[tid] = 0.f;
  __syncthreads();
  float* qraw = ws + OFF_QRAW;
  float* kraw = ws + OFF_KRAW;
  float* vraw = ws + OFF_VRAW;
  for (int i = 0; i < 14; ++i) {             // 112 out-ch * 32 n / 256 thr
    int idx = tid + i * 256;
    int o = idx >> 5, j = idx & 31;
    const float* wrow = (o < 64) ? (wq + o * CIN)
                      : (o < 80) ? (wk + (o - 64) * CIN)
                                 : (wv + (o - 80) * CIN);
    float acc = 0.f;
#pragma unroll 8
    for (int c = 0; c < CIN; ++c) acc += wrow[c] * xs[c][j];
    if (o < 64)      qraw[((b * 64 + o)        << 10) + n0 + j] = acc;
    else if (o < 80) kraw[((b * 16 + (o - 64)) << 10) + n0 + j] = acc;
    else             vraw[((b * 32 + (o - 80)) << 10) + n0 + j] = acc;
    if (o < 64 || o >= 80) {
      atomicAdd(&st[o], acc);
      atomicAdd(&st[112 + o], acc * acc);
    }
  }
  __syncthreads();
  if (tid < 224 && ((tid % 112) < 64 || (tid % 112) >= 80))
    atomicAdd(&ws[OFF_STAT + tid], st[tid]);
}

__global__ void finalize_kernel(const float* __restrict__ gq,
                                const float* __restrict__ bq,
                                const float* __restrict__ gv,
                                const float* __restrict__ bv, float* ws) {
  int t = threadIdx.x;
  const float invN = 1.f / (float)(BATCH * NPIX);
  float* sc = ws + OFF_SCSH;
  if (t < 64) {
    float s = ws[OFF_STAT + t], sq = ws[OFF_STAT + 112 + t];
    float mean = s * invN;
    float var = sq * invN - mean * mean;
    float a = gq[t] * rsqrtf(var + EPS);
    sc[t] = a; sc[64 + t] = bq[t] - mean * a;
  } else if (t < 96) {
    int c = t - 64;
    float s = ws[OFF_STAT + 80 + c], sq = ws[OFF_STAT + 112 + 80 + c];
    float mean = s * invN;
    float var = sq * invN - mean * mean;
    float a = gv[c] * rsqrtf(var + EPS);
    sc[128 + c] = a; sc[160 + c] = bv[c] - mean * a;
  }
}

// softmax(k) fused into lam_c[b][k][v] = sum_m softmax(k)[m] * vbar[v][m]
__global__ __launch_bounds__(256) void lamc_kernel(float* ws) {
  __shared__ float sm[1024];
  __shared__ float red[256];
  __shared__ float lc[32];
  __shared__ float inv_s;
  const int k = blockIdx.x, b = blockIdx.y, tid = threadIdx.x;
  const float* row = ws + OFF_KRAW + ((b * 16 + k) << 10);
  float mx = -3.4e38f;
  for (int j = 0; j < 4; ++j) mx = fmaxf(mx, row[tid + (j << 8)]);
  red[tid] = mx; __syncthreads();
  for (int s = 128; s > 0; s >>= 1) {
    if (tid < s) red[tid] = fmaxf(red[tid], red[tid + s]);
    __syncthreads();
  }
  mx = red[0]; __syncthreads();
  float sum = 0.f;
  for (int j = 0; j < 4; ++j) {
    int m = tid + (j << 8);
    float e = __expf(row[m] - mx);
    sm[m] = e; sum += e;
  }
  red[tid] = sum; __syncthreads();
  for (int s = 128; s > 0; s >>= 1) {
    if (tid < s) red[tid] += red[tid + s];
    __syncthreads();
  }
  if (tid == 0) inv_s = 1.f / red[0];
  if (tid < 32) lc[tid] = 0.f;
  __syncthreads();
  const float* scv = ws + OFF_SCSH + 128;
  const float* shv = ws + OFF_SCSH + 160;
  const float* vraw = ws + OFF_VRAW + ((b * 32) << 10);
  float p[32];
#pragma unroll
  for (int v = 0; v < 32; ++v) p[v] = 0.f;
  for (int j = 0; j < 4; ++j) {
    int m = tid + (j << 8);
    float s = sm[m];
#pragma unroll
    for (int v = 0; v < 32; ++v)
      p[v] += s * (vraw[(v << 10) + m] * scv[v] + shv[v]);
  }
#pragma unroll
  for (int v = 0; v < 32; ++v) atomicAdd(&lc[v], p[v]);
  __syncthreads();
  if (tid < 32) ws[OFF_LAMC + (b * 16 + k) * 32 + tid] = lc[tid] * inv_s;
}

// vbar (BN applied) -> bf16 [b][v][m]
__global__ void vb16_kernel(float* ws) {
  int idx = blockIdx.x * 256 + threadIdx.x;   // 524288 total
  const float* scv = ws + OFF_SCSH + 128;
  const float* shv = ws + OFF_SCSH + 160;
  int v = (idx >> 10) & 31;
  float val = ws[OFF_VRAW + idx] * scv[v] + shv[v];
  ((unsigned short*)(ws + OFF_VB16))[idx] = f2bf(val);
}

// pos_emb [n][m][k] fp32 -> Pt [n][k][m] bf16 (LDS-tiled transpose)
__global__ __launch_bounds__(256) void pt_kernel(const float* __restrict__ pos,
                                                 float* ws) {
  __shared__ unsigned short t[16][72];
  const int n = blockIdx.x, tid = threadIdx.x;
  unsigned short* Pt = (unsigned short*)(ws + OFF_PT) + n * (16 * NPIX);
  const float* Pn = pos + n * (NPIX * 16);
  for (int chunk = 0; chunk < 16; ++chunk) {
    int m0 = chunk * 64;
    for (int pass = 0; pass < 4; ++pass) {
      int mm = (tid >> 4) + pass * 16;
      int kk = tid & 15;
      t[kk][mm] = f2bf(Pn[(m0 + mm) * 16 + kk]);
    }
    __syncthreads();
    for (int pass = 0; pass < 4; ++pass) {
      int i = tid + pass * 256;
      int k = i >> 6, m = i & 63;
      Pt[k * NPIX + m0 + m] = t[k][m];
    }
    __syncthreads();
  }
}

union FragBf { uint4 u[2]; v16bf v; };

// main: per-(b,n) lam_p via bf16 WMMA, fused +lam_c and Y = q * lam
__global__ __launch_bounds__(512) void lamp_y_kernel(float* __restrict__ ws,
                                                     float* __restrict__ out) {
  __shared__ float q_s[16][64];
  __shared__ float lam_s[16][16][32];
  __shared__ float ybuf[128][17];
  const int b = blockIdx.y;
  const int n0 = blockIdx.x * 16;
  const int tid = threadIdx.x;
  const int w = tid >> 5;       // wave id -> n offset in tile
  const int lane = tid & 31;
  const int lk = lane & 15;
  const int hi = lane >> 4;
  const int n = n0 + w;
  {  // stage BN'd q row for this n
    const float* qraw = ws + OFF_QRAW + ((b * 64) << 10);
    const float* scq = ws + OFF_SCSH;
    const float* shq = ws + OFF_SCSH + 64;
    int c0i = lane, c1i = lane + 32;
    q_s[w][c0i] = qraw[(c0i << 10) + n] * scq[c0i] + shq[c0i];
    q_s[w][c1i] = qraw[(c1i << 10) + n] * scq[c1i] + shq[c1i];
  }
  // A = Pt[n] : [k][m] bf16 ; B = vb16[b] : [v][m] bf16
  const unsigned short* A  = (const unsigned short*)(ws + OFF_PT) + n * (16 * 1024) + lk * 1024;
  const unsigned short* B0 = (const unsigned short*)(ws + OFF_VB16) + b * (32 * 1024) + lk * 1024;
  const unsigned short* B1 = B0 + 16 * 1024;
  v8f c0 = {}; v8f c1 = {};
  for (int m0 = 0; m0 < 1024; m0 += 32) {
    FragBf a, b0, b1;
    // A k order used here (contiguous, not the ISA table order): lanes0-15 K=0..15, lanes16-31 K=16..31 (matches B)
    a.u[0]  = *(const uint4*)(A  + m0 + hi * 16);
    a.u[1]  = *(const uint4*)(A  + m0 + hi * 16 + 8);
    // B k order used here (contiguous, same as A): lane holds col v; lanes0-15 K=0..15, lanes16-31 K=16..31
    b0.u[0] = *(const uint4*)(B0 + m0 + hi * 16);
    b0.u[1] = *(const uint4*)(B0 + m0 + hi * 16 + 8);
    b1.u[0] = *(const uint4*)(B1 + m0 + hi * 16);
    b1.u[1] = *(const uint4*)(B1 + m0 + hi * 16 + 8);
    __builtin_prefetch((const void*)(A + m0 + 64), 0, 0);
    c0 = __builtin_amdgcn_wmma_f32_16x16x32_bf16(false, a.v, false, b0.v,
                                                 (short)0, c0, false, false);
    c1 = __builtin_amdgcn_wmma_f32_16x16x32_bf16(false, a.v, false, b1.v,
                                                 (short)0, c1, false, false);
  }
  // C/D layout: VGPR r, lane L -> M=r+8*(L>=16), N=L%16.  M=k, N=v.
  const float* lamcB = ws + OFF_LAMC + (b * 16) * 32;
#pragma unroll
  for (int r = 0; r < 8; ++r) {
    int k = r + 8 * hi;
    lam_s[w][k][lk]      = c0[r] + lamcB[k * 32 + lk];
    lam_s[w][k][lk + 16] = c1[r] + lamcB[k * 32 + lk + 16];
  }
  __syncthreads();
  {  // Y[h][v] = sum_k q[h*16+k] * lam[k][v]; lane owns column v
    int v = lane;
#pragma unroll
    for (int h = 0; h < 4; ++h) {
      float y = 0.f;
#pragma unroll
      for (int k = 0; k < 16; ++k) y += q_s[w][h * 16 + k] * lam_s[w][k][v];
      ybuf[h * 32 + v][w] = y;
    }
  }
  __syncthreads();
  for (int i = tid; i < 2048; i += 512) {   // coalesced along n
    int ch = i >> 4, j = i & 15;
    out[((b * 128 + ch) << 10) + n0 + j] = ybuf[ch][j];
  }
}

extern "C" void kernel_launch(void* const* d_in, const int* in_sizes, int n_in,
                              void* d_out, int out_size, void* d_ws, size_t ws_size,
                              hipStream_t stream) {
  const float* x   = (const float*)d_in[0];
  const float* wq  = (const float*)d_in[1];
  const float* wk  = (const float*)d_in[2];
  const float* wv  = (const float*)d_in[3];
  const float* pos = (const float*)d_in[4];
  const float* gq  = (const float*)d_in[5];
  const float* bq  = (const float*)d_in[6];
  const float* gv  = (const float*)d_in[7];
  const float* bv  = (const float*)d_in[8];
  float* ws  = (float*)d_ws;
  float* out = (float*)d_out;

  zero_stats_kernel<<<1, 256, 0, stream>>>(ws);
  proj_stats_kernel<<<dim3(32, 16), 256, 0, stream>>>(x, wq, wk, wv, ws);
  finalize_kernel<<<1, 128, 0, stream>>>(gq, bq, gv, bv, ws);
  lamc_kernel<<<dim3(16, 16), 256, 0, stream>>>(ws);
  vb16_kernel<<<2048, 256, 0, stream>>>(ws);
  pt_kernel<<<1024, 256, 0, stream>>>(pos, ws);
  lamp_y_kernel<<<dim3(64, 16), 512, 0, stream>>>(ws, out);
}
